// YOWO_33998961115397
// MI455X (gfx1250) — compile-verified, matrix operand order checked by reading, not run on hardware
//
#include <hip/hip_runtime.h>
#include <cstdint>

// YOLO-style postprocess for MI455X (gfx1250).
// Stage A: async global->LDS streaming of cls logits + tournament top-50 per 8192-chunk.
// Stage B: per (image,level) tournament merge of chunk candidates + box decode.
// Stage C: rank-sort of 150 dets via v_wmma_f32_16x16x32_f16 (comparison-matrix row sums),
//          then class-offset greedy NMS, write [B,150,6].
// Workspace: candScore f32[B*329*50] | candIdx i32[B*329*50] | detBuf f32[B*150*6]  (~4.33 MB)

#define NCLS 80
#define TOPK 50
#define KTOT 150
#define CHUNK 8192
#define NTHR 256
#define NEG (-1.0e30f)
#define CHUNKS_TOT 329   // 250 + 63 + 16 chunks per image
#define NBATCH 32

typedef _Float16 v16h __attribute__((ext_vector_type(16)));
typedef float    v8f  __attribute__((ext_vector_type(8)));

__device__ __forceinline__ float sigmoidf_(float x) { return 1.0f / (1.0f + __expf(-x)); }

// CDNA5 async global->LDS copy (ASYNCcnt-tracked), 16B per lane per issue.
__device__ __forceinline__ void async_load_b128(unsigned lds_off, const void* g) {
    asm volatile("global_load_async_to_lds_b128 %0, %1, off"
                 :: "v"(lds_off), "v"((unsigned long long)(uintptr_t)g)
                 : "memory");
}
__device__ __forceinline__ void wait_async0() {
    asm volatile("s_wait_asynccnt 0" ::: "memory");
}

// ---------------- Stage A: per-chunk score compute + top-50 --------------------------
__global__ __launch_bounds__(NTHR)
void topk_chunks(const float* __restrict__ conf, const float* __restrict__ cls,
                 int M, int N, int candBase,
                 float* __restrict__ candScore, int* __restrict__ candIdx)
{
    __shared__ __align__(16) float sScore[CHUNK];   // 32 KB: raw cls, then scores in place
    __shared__ float sSigConf[112];
    __shared__ float rv[8]; __shared__ int rg[8]; __shared__ int rt[8];
    __shared__ int sWin;

    const int t    = threadIdx.x;
    const int b    = blockIdx.y;
    const int base = blockIdx.x * CHUNK;
    const float* clsB = cls + (size_t)b * N;

    // 1) async stream the chunk's class logits into LDS (tail handled with guarded scalar path)
#pragma unroll
    for (int k = 0; k < CHUNK / 4 / NTHR; ++k) {
        const int e0 = 4 * (t + k * NTHR);
        const int g0 = base + e0;
        if (g0 + 4 <= N) {
            async_load_b128((unsigned)(size_t)&sScore[e0], clsB + g0);
        } else {
            for (int e = 0; e < 4; ++e) {
                const int gi = g0 + e;
                sScore[e0 + e] = (gi < N) ? clsB[gi] : 0.0f;
            }
        }
    }
    // 2) sigmoid(conf) for the <=104 anchors this chunk touches
    const int nEnd = min(base + CHUNK, N);
    const int a0   = base / NCLS;
    const int aCnt = (nEnd - 1) / NCLS - a0 + 1;
    for (int i = t; i < aCnt; i += NTHR)
        sSigConf[i] = sigmoidf_(conf[(size_t)b * M + a0 + i]);
    wait_async0();
    __syncthreads();

    // 3) scores in place; thread t owns interleaved slots {t + 256j}
#pragma unroll
    for (int j = 0; j < CHUNK / NTHR; ++j) {
        const int i  = t + (j << 8);
        const int gi = base + i;
        float s = NEG;
        if (gi < N) {
            const float sc = sigmoidf_(sScore[i]);
            const int a = gi / NCLS;
            s = sqrtf(sSigConf[a - a0] * sc);
        }
        sScore[i] = s;
    }
    // no barrier needed: each thread only reads its own slots below

    // 4) tournament top-50: register heads + wave-shuffle reduce; only the winner rescans
    float bv = NEG; int bg = 0x7fffffff; int bj = 0;
#pragma unroll
    for (int j = 0; j < CHUNK / NTHR; ++j) {
        const int idx = t + (j << 8);
        const float v = sScore[idx];
        const int   g = base + idx;
        if (v > bv || (v == bv && g < bg)) { bv = v; bg = g; bj = j; }
    }
    float* outS = candScore + ((size_t)b * CHUNKS_TOT + candBase + blockIdx.x) * TOPK;
    int*   outI = candIdx   + ((size_t)b * CHUNKS_TOT + candBase + blockIdx.x) * TOPK;

    for (int sel = 0; sel < TOPK; ++sel) {
        float v = bv; int g = bg; int wt = t;
#pragma unroll
        for (int off = 16; off > 0; off >>= 1) {
            const float ov = __shfl_down(v, off);
            const int   og = __shfl_down(g, off);
            const int   ot = __shfl_down(wt, off);
            if (ov > v || (ov == v && og < g)) { v = ov; g = og; wt = ot; }
        }
        if ((t & 31) == 0) { rv[t >> 5] = v; rg[t >> 5] = g; rt[t >> 5] = wt; }
        __syncthreads();
        if (t == 0) {
            float fv = rv[0]; int fg = rg[0], ft = rt[0];
            for (int w = 1; w < 8; ++w)
                if (rv[w] > fv || (rv[w] == fv && rg[w] < fg)) { fv = rv[w]; fg = rg[w]; ft = rt[w]; }
            outS[sel] = fv; outI[sel] = fg; sWin = ft;
        }
        __syncthreads();
        if (t == sWin) {
            sScore[t + (bj << 8)] = NEG;
            bv = NEG; bg = 0x7fffffff; bj = 0;
#pragma unroll
            for (int j = 0; j < CHUNK / NTHR; ++j) {
                const int idx = t + (j << 8);
                const float vv = sScore[idx];
                const int   gg = base + idx;
                if (vv > bv || (vv == bv && gg < bg)) { bv = vv; bg = gg; bj = j; }
            }
        }
        __syncthreads();
    }
}

// ---------------- Stage B: merge chunk candidates per (image, level) + decode -------
__global__ __launch_bounds__(NTHR)
void merge_decode(const float* __restrict__ reg0, const float* __restrict__ reg1,
                  const float* __restrict__ reg2,
                  float* __restrict__ candScore, const int* __restrict__ candIdx,
                  float* __restrict__ detBuf)
{
    __shared__ float rv[8]; __shared__ int rg[8]; __shared__ int rt[8];
    __shared__ int sWin;
    const int t   = threadIdx.x;
    const int lvl = blockIdx.x;
    const int b   = blockIdx.y;
    const int   chunksL[3] = {250, 63, 16};
    const int   candB[3]   = {0, 250, 313};
    const int   Ml[3]      = {25600, 6400, 1600};
    const int   Wl[3]      = {160, 80, 40};
    const float strl[3]    = {8.f, 16.f, 32.f};
    const float* regp = (lvl == 0) ? reg0 : (lvl == 1) ? reg1 : reg2;
    const int nC = chunksL[lvl] * TOPK;
    float*      csc = candScore + ((size_t)b * CHUNKS_TOT + candB[lvl]) * TOPK;
    const int*  cix = candIdx   + ((size_t)b * CHUNKS_TOT + candB[lvl]) * TOPK;
    const int M = Ml[lvl], W = Wl[lvl];
    const float stride = strl[lvl];

    float bv = NEG; int bg = 0x7fffffff; int bp = 0;
    for (int i = t; i < nC; i += NTHR) {
        const float v = csc[i]; const int g = cix[i];
        if (v > bv || (v == bv && g < bg)) { bv = v; bg = g; bp = i; }
    }
    for (int sel = 0; sel < TOPK; ++sel) {
        float v = bv; int g = bg; int wt = t;
#pragma unroll
        for (int off = 16; off > 0; off >>= 1) {
            const float ov = __shfl_down(v, off);
            const int   og = __shfl_down(g, off);
            const int   ot = __shfl_down(wt, off);
            if (ov > v || (ov == v && og < g)) { v = ov; g = og; wt = ot; }
        }
        if ((t & 31) == 0) { rv[t >> 5] = v; rg[t >> 5] = g; rt[t >> 5] = wt; }
        __syncthreads();
        if (t == 0) {
            float fv = rv[0]; int fg = rg[0], ft = rt[0];
            for (int w = 1; w < 8; ++w)
                if (rv[w] > fv || (rv[w] == fv && rg[w] < fg)) { fv = rv[w]; fg = rg[w]; ft = rt[w]; }
            sWin = ft;
            // decode detection sel: gi -> (anchor, class) -> box
            const int gi  = fg;
            const int a   = gi / NCLS;
            const int lab = gi - a * NCLS;
            const float ax = ((float)(a % W) + 0.5f) * stride;
            const float ay = ((float)(a / W) + 0.5f) * stride;
            const float* rp = regp + ((size_t)b * M + a) * 4;
            const float cx = ax + rp[0] * stride;
            const float cy = ay + rp[1] * stride;
            const float hw = 0.5f * __expf(rp[2]) * stride;
            const float hh = 0.5f * __expf(rp[3]) * stride;
            const float s  = (fv > 0.05f) ? fv : 0.0f;
            float* o = detBuf + ((size_t)b * KTOT + lvl * TOPK + sel) * 6;
            o[0] = cx - hw; o[1] = cy - hh; o[2] = cx + hw; o[3] = cy + hh;
            o[4] = s; o[5] = (float)lab;
        }
        __syncthreads();
        if (t == sWin) {
            csc[bp] = NEG;
            bv = NEG; bg = 0x7fffffff; bp = 0;
            for (int i = t; i < nC; i += NTHR) {
                const float vv = csc[i]; const int gg = cix[i];
                if (vv > bv || (vv == bv && gg < bg)) { bv = vv; bg = gg; bp = i; }
            }
        }
        __syncthreads();
    }
}

// ---------------- Stage C: WMMA rank-sort + class-offset NMS + output ---------------
__global__ __launch_bounds__(NTHR)
void sort_nms_out(const float* __restrict__ detBuf, float* __restrict__ out)
{
    __shared__ float sS[160], sLab[160], sBox[160][4];
    __shared__ int   sRank[160];
    __shared__ float qS[160], qLab[160], qBox[160][4];
    __shared__ int   keepF[160];
    const int t = threadIdx.x;
    const int b = blockIdx.x;

    for (int i = t; i < 160; i += NTHR) {
        if (i < KTOT) {
            const float* d = detBuf + ((size_t)b * KTOT + i) * 6;
            sBox[i][0] = d[0]; sBox[i][1] = d[1]; sBox[i][2] = d[2]; sBox[i][3] = d[3];
            sS[i] = d[4]; sLab[i] = d[5];
        } else {
            sS[i] = NEG; sLab[i] = 0.f;
            sBox[i][0] = sBox[i][1] = sBox[i][2] = sBox[i][3] = 0.f;
        }
    }
    __syncthreads();

    // rank[m] = #{k : s[k] > s[m] || (s[k]==s[m] && k<m)} as row-sums of a 0/1 matrix,
    // computed on the matrix unit: A (16x32 f16 fragments) x ones(32x16) accumulated in f32.
    if (t < 32) {
        const int lane = t, lh = lane >> 4, mrow = lane & 15;
        v16h ones;
#pragma unroll
        for (int e = 0; e < 16; ++e) ones[e] = (_Float16)1.0f;
#pragma unroll
        for (int mt = 0; mt < 10; ++mt) {
            const int m = mt * 16 + mrow;
            const float sm = sS[m];
            v8f acc = {0.f, 0.f, 0.f, 0.f, 0.f, 0.f, 0.f, 0.f};
#pragma unroll
            for (int kt = 0; kt < 5; ++kt) {
                v16h a;
#pragma unroll
                for (int e = 0; e < 16; ++e) {
                    // 16-bit A 16x32 fragment layout: K(e) = 16*(e>>3) + 8*(lane>>4) + (e&7)
                    const int k = kt * 32 + ((e >> 3) << 4) + (lh << 3) + (e & 7);
                    const float sk = sS[k];
                    a[e] = (_Float16)((sk > sm || (sk == sm && k < m)) ? 1.0f : 0.0f);
                }
                acc = __builtin_amdgcn_wmma_f32_16x16x32_f16(false, a, false, ones,
                                                             (short)0, acc, false, false);
            }
            // D layout: VGPR r -> M = r + 8*(lane>>4); all N columns equal (ones-B)
            if (mrow == 0) {
#pragma unroll
                for (int r = 0; r < 8; ++r)
                    sRank[mt * 16 + (lh << 3) + r] = (int)(acc[r] + 0.5f);
            }
        }
    }
    __syncthreads();

    for (int i = t; i < 160; i += NTHR) {
        int r = sRank[i];
        r = (r < 0) ? 0 : (r > 159 ? 159 : r);
        qS[r] = sS[i]; qLab[r] = sLab[i];
        qBox[r][0] = sBox[i][0]; qBox[r][1] = sBox[i][1];
        qBox[r][2] = sBox[i][2]; qBox[r][3] = sBox[i][3];
    }
    __syncthreads();
    for (int i = t; i < 160; i += NTHR) keepF[i] = (i < KTOT) && (qS[i] > 0.0f);

    // greedy class-offset NMS (sequential over kept i, parallel over j)
    for (int i = 0; i < KTOT; ++i) {
        __syncthreads();
        if (!keepF[i]) continue;   // uniform: all threads read the same LDS value
        const float ofi = qLab[i] * 100000.0f;
        const float x1 = qBox[i][0] + ofi, y1 = qBox[i][1] + ofi;
        const float x2 = qBox[i][2] + ofi, y2 = qBox[i][3] + ofi;
        const float ai = fmaxf(x2 - x1, 0.f) * fmaxf(y2 - y1, 0.f);
        for (int j = i + 1 + t; j < KTOT; j += NTHR) {
            if (!keepF[j]) continue;
            const float ofj = qLab[j] * 100000.0f;
            const float u1 = qBox[j][0] + ofj, v1 = qBox[j][1] + ofj;
            const float u2 = qBox[j][2] + ofj, v2 = qBox[j][3] + ofj;
            const float iw = fmaxf(fminf(x2, u2) - fmaxf(x1, u1), 0.f);
            const float ih = fmaxf(fminf(y2, v2) - fmaxf(y1, v1), 0.f);
            const float inter = iw * ih;
            const float aj = fmaxf(u2 - u1, 0.f) * fmaxf(v2 - v1, 0.f);
            if (inter / (ai + aj - inter + 1e-9f) > 0.6f) keepF[j] = 0;
        }
    }
    __syncthreads();

    for (int i = t; i < KTOT; i += NTHR) {
        float* o = out + ((size_t)b * KTOT + i) * 6;
        const bool k = keepF[i] != 0;
        o[0] = k ? qBox[i][0] : 0.f; o[1] = k ? qBox[i][1] : 0.f;
        o[2] = k ? qBox[i][2] : 0.f; o[3] = k ? qBox[i][3] : 0.f;
        o[4] = k ? qS[i] : 0.f;      o[5] = qLab[i];
    }
}

// ------------------------------------------------------------------------------------
extern "C" void kernel_launch(void* const* d_in, const int* in_sizes, int n_in,
                              void* d_out, int out_size, void* d_ws, size_t ws_size,
                              hipStream_t stream)
{
    (void)in_sizes; (void)n_in; (void)out_size; (void)ws_size;
    const float* conf0 = (const float*)d_in[0];
    const float* cls0  = (const float*)d_in[1];
    const float* reg0  = (const float*)d_in[2];
    const float* conf1 = (const float*)d_in[3];
    const float* cls1  = (const float*)d_in[4];
    const float* reg1  = (const float*)d_in[5];
    const float* conf2 = (const float*)d_in[6];
    const float* cls2  = (const float*)d_in[7];
    const float* reg2  = (const float*)d_in[8];
    float* out = (float*)d_out;

    char* ws = (char*)d_ws;
    float* candScore = (float*)ws;                                              // B*329*50 f32
    int*   candIdx   = (int*)  (ws + (size_t)4 * NBATCH * CHUNKS_TOT * TOPK);   // B*329*50 i32
    float* detBuf    = (float*)(ws + (size_t)8 * NBATCH * CHUNKS_TOT * TOPK);   // B*150*6 f32

    topk_chunks<<<dim3(250, NBATCH), NTHR, 0, stream>>>(conf0, cls0, 25600, 25600 * NCLS, 0,   candScore, candIdx);
    topk_chunks<<<dim3(63,  NBATCH), NTHR, 0, stream>>>(conf1, cls1, 6400,  6400  * NCLS, 250, candScore, candIdx);
    topk_chunks<<<dim3(16,  NBATCH), NTHR, 0, stream>>>(conf2, cls2, 1600,  1600  * NCLS, 313, candScore, candIdx);
    merge_decode<<<dim3(3, NBATCH), NTHR, 0, stream>>>(reg0, reg1, reg2, candScore, candIdx, detBuf);
    sort_nms_out<<<NBATCH, NTHR, 0, stream>>>(detBuf, out);
}
